// NoisyPositionEmbedder_80917183857335
// MI455X (gfx1250) — matrix operand order repaired and checked by reading, not compiled, not run on hardware
//
#include <hip/hip_runtime.h>
#include <stdint.h>

// ---------------------------------------------------------------------------
// Types for the CDNA5 WMMA builtin (codegen-confirmed signature):
//   v8f = __builtin_amdgcn_wmma_f32_16x16x32_bf16(bool, v16bf, bool, v16bf,
//                                                 short, v8f, bool, bool)
// ---------------------------------------------------------------------------
typedef __attribute__((ext_vector_type(16))) __bf16 v16bf;
typedef __attribute__((ext_vector_type(8)))  float  v8f;

union BF16x16 { v16bf v; uint32_t u[8]; };

// round-to-nearest-even f32 -> bf16, packed pair (low 16 bits = first elem)
__device__ __forceinline__ uint32_t pack_bf16x2(float lo, float hi) {
    union { float f; uint32_t u; } a, b;
    a.f = lo; b.f = hi;
    uint32_t ua = a.u + 0x7FFFu + ((a.u >> 16) & 1u);
    uint32_t ub = b.u + 0x7FFFu + ((b.u >> 16) & 1u);
    return (ua >> 16) | (ub & 0xFFFF0000u);
}

#define C_Z   128
#define C_P   16
#define C_S   384
#define C_A   128
#define N_TOK 256
#define N_AT  1024

// ---------------------------------------------------------------------------
// Kernel A: P[ti*256+tj, 0:16] = LN(zij[ti,tj,:]) @ w_z   (65536 rows, K=128)
// One wave per 16-row x 16-col tile; LN fused via LDS + cross-half shuffle.
// ---------------------------------------------------------------------------
__global__ __launch_bounds__(128) void pair_ln_gemm_kernel(
    const float* __restrict__ zij,              // [256,256,128]
    const float* __restrict__ lnw,              // [128]
    const float* __restrict__ lnb,              // [128]
    const float* __restrict__ wz,               // [128,16]
    float* __restrict__ P)                      // [65536,16]
{
    __shared__ float zbuf[4][16 * 132];         // 132-float row stride (pad)
    const int wave = threadIdx.x >> 5;
    const int lane = threadIdx.x & 31;
    const int tile = blockIdx.x * 4 + wave;     // 0..4095
    const int p0   = tile * 16;                 // first token-pair row
    float* zb = zbuf[wave];

    // Cooperative load: 16 rows x 128 f32 = 512 float4 (coalesced)
    const float4* src = (const float4*)(zij + (size_t)p0 * C_Z);
#pragma unroll
    for (int i = 0; i < 16; ++i) {
        int e4  = i * 32 + lane;                // 0..511
        int row = e4 >> 5;
        int c4  = e4 & 31;
        *(float4*)(zb + row * 132 + c4 * 4) = src[e4];
    }
    __syncthreads();

    const int row = lane & 15;                  // A-matrix row for this lane
    const int lh  = lane >> 4;                  // lane half (K-subrange sel)
    const int n   = lane & 15;                  // B/D column

    // LN statistics: each lane sums half of its row, combine across halves.
    float s1 = 0.f, s2 = 0.f;
    const float* zr = zb + row * 132 + lh * 64;
#pragma unroll 8
    for (int c = 0; c < 64; ++c) { float x = zr[c]; s1 += x; s2 += x * x; }
    s1 += __shfl_xor(s1, 16, 32);
    s2 += __shfl_xor(s2, 16, 32);
    const float mu  = s1 * (1.f / 128.f);
    const float var = s2 * (1.f / 128.f) - mu * mu;
    const float rs  = rsqrtf(var + 1e-5f);

    // B fragments (w_z), dense 16-bit B 32x16 layout:
    // lane: N = lane&15; VGPR v holds K = 32t + 16*(v>>2) + 8*lh + 2*(v&3), +1
    BF16x16 Bf[4];
#pragma unroll
    for (int t = 0; t < 4; ++t)
#pragma unroll
        for (int v = 0; v < 8; ++v) {
            int k = t * 32 + ((v >> 2) << 4) + lh * 8 + ((v & 3) << 1);
            Bf[t].u[v] = pack_bf16x2(wz[k * C_P + n], wz[(k + 1) * C_P + n]);
        }

    // A fragments + 4 chained WMMAs over K=128
    v8f acc = {};
    const float* zrow = zb + row * 132;
#pragma unroll
    for (int t = 0; t < 4; ++t) {
        BF16x16 A;
#pragma unroll
        for (int v = 0; v < 8; ++v) {
            // 16-bit A 16x32 layout: K = 32t + 16*(v>>2) + 8*lh + 2*(v&3)
            int k = t * 32 + ((v >> 2) << 4) + lh * 8 + ((v & 3) << 1);
            float a0 = (zrow[k]     - mu) * rs * lnw[k]     + lnb[k];
            float a1 = (zrow[k + 1] - mu) * rs * lnw[k + 1] + lnb[k + 1];
            A.u[v] = pack_bf16x2(a0, a1);
        }
        acc = __builtin_amdgcn_wmma_f32_16x16x32_bf16(
            false, A.v, false, Bf[t].v, (short)0, acc, false, false);
    }

    // D layout: VGPR r -> (M = r + 8*lh, N = lane&15)
    float* dst = P + (size_t)p0 * C_P + n;
#pragma unroll
    for (int r = 0; r < 8; ++r)
        dst[(size_t)(r + lh * 8) * C_P] = acc[r];
}

// ---------------------------------------------------------------------------
// Kernel C1: LN(si_trunk) -> packed bf16 pairs [256, 192] (K pairs of 384)
// ---------------------------------------------------------------------------
__global__ __launch_bounds__(128) void ln_si_kernel(
    const float* __restrict__ si,               // [256,384]
    const float* __restrict__ lnw,
    const float* __restrict__ lnb,
    uint32_t* __restrict__ out)                 // [256,192]
{
    __shared__ float r1[128], r2[128];
    const int ti = blockIdx.x;
    const int t  = threadIdx.x;
    const float* row = si + (size_t)ti * C_S;
    float s1 = 0.f, s2 = 0.f;
#pragma unroll
    for (int j = 0; j < 3; ++j) { float x = row[t + 128 * j]; s1 += x; s2 += x * x; }
    r1[t] = s1; r2[t] = s2;
    __syncthreads();
    for (int off = 64; off > 0; off >>= 1) {
        if (t < off) { r1[t] += r1[t + off]; r2[t] += r2[t + off]; }
        __syncthreads();
    }
    const float mu  = r1[0] * (1.f / 384.f);
    const float var = r2[0] * (1.f / 384.f) - mu * mu;
    const float rs  = rsqrtf(var + 1e-5f);
    if (t < 96) {
#pragma unroll
        for (int q = 0; q < 2; ++q) {
            int k = t * 4 + q * 2;
            float a0 = (row[k]     - mu) * rs * lnw[k]     + lnb[k];
            float a1 = (row[k + 1] - mu) * rs * lnw[k + 1] + lnb[k + 1];
            out[ti * 192 + (k >> 1)] = pack_bf16x2(a0, a1);
        }
    }
}

// ---------------------------------------------------------------------------
// Kernel C2: S[256,128] = LN(si) @ w_s  (K=384 -> 12 WMMA chunks)
// 128 waves: 16 row tiles x 8 col tiles
// ---------------------------------------------------------------------------
__global__ __launch_bounds__(128) void s_gemm_kernel(
    const uint32_t* __restrict__ lnsi,          // [256,192] bf16 pairs
    const float* __restrict__ w_s,              // [384,128]
    float* __restrict__ S)                      // [256,128]
{
    const int wave = threadIdx.x >> 5;
    const int lane = threadIdx.x & 31;
    const int wg   = blockIdx.x * 4 + wave;     // 0..127
    const int rt = wg >> 3, ct = wg & 7;
    const int n  = lane & 15, lh = lane >> 4;
    const int ng = ct * 16 + n;
    const int rowg = rt * 16 + n;               // A row for this lane (=lane&15)

    v8f acc = {};
#pragma unroll
    for (int t = 0; t < 12; ++t) {
        BF16x16 A, B;
#pragma unroll
        for (int v = 0; v < 8; ++v) {
            int ka = t * 32 + ((v >> 2) << 4) + lh * 8 + ((v & 3) << 1);
            A.u[v] = lnsi[rowg * 192 + (ka >> 1)];
            int kb = t * 32 + ((v >> 2) << 4) + lh * 8 + ((v & 3) << 1);
            B.u[v] = pack_bf16x2(w_s[kb * C_A + ng], w_s[(kb + 1) * C_A + ng]);
        }
        acc = __builtin_amdgcn_wmma_f32_16x16x32_bf16(
            false, A.v, false, B.v, (short)0, acc, false, false);
    }
#pragma unroll
    for (int r = 0; r < 8; ++r)
        S[(size_t)(rt * 16 + r + lh * 8) * C_A + ng] = acc[r];
}

// ---------------------------------------------------------------------------
// Kernel D: plm_out = plm + P[tok[l],tok[m],:]   (pure streaming, 128 MB)
// One float4 per thread; P is 4 MB -> L2 resident.
// ---------------------------------------------------------------------------
__global__ __launch_bounds__(256) void plm_add_kernel(
    const float4* __restrict__ plm4,            // [1024*1024*4]
    const int*    __restrict__ tok,             // [1024]
    const float4* __restrict__ P4,              // [65536*4]
    float4*       __restrict__ out4)
{
    size_t i = (size_t)blockIdx.x * blockDim.x + threadIdx.x; // < 4194304
    int c4 = (int)(i & 3);
    int m  = (int)((i >> 2) & 1023);
    int l  = (int)(i >> 12);
    float4 a = plm4[i];
    int ti = tok[l], tj = tok[m];
    float4 b = P4[((size_t)(ti * N_TOK + tj) << 2) + c4];
    float4 o; o.x = a.x + b.x; o.y = a.y + b.y; o.z = a.z + b.z; o.w = a.w + b.w;
    out4[i] = o;
}

// ---------------------------------------------------------------------------
// Kernel E: cl_out = cl + S[tok[l],:]  and  ql_out = ql + rl @ w_r
// ---------------------------------------------------------------------------
__global__ __launch_bounds__(256) void small_out_kernel(
    const float* __restrict__ cl, const float* __restrict__ S,
    const int*   __restrict__ tok,
    const float* __restrict__ ql, const float* __restrict__ rl,
    const float* __restrict__ wr,               // [3,128]
    float* __restrict__ cl_out, float* __restrict__ ql_out)
{
    int i = blockIdx.x * blockDim.x + threadIdx.x; // < 131072
    int l = i >> 7, c = i & 127;
    cl_out[i] = cl[i] + S[tok[l] * C_A + c];
    float r0 = rl[l * 3 + 0], r1 = rl[l * 3 + 1], r2 = rl[l * 3 + 2];
    ql_out[i] = ql[i] + r0 * wr[c] + r1 * wr[128 + c] + r2 * wr[256 + c];
}

// ---------------------------------------------------------------------------
extern "C" void kernel_launch(void* const* d_in, const int* in_sizes, int n_in,
                              void* d_out, int out_size, void* d_ws, size_t ws_size,
                              hipStream_t stream) {
    const int*   tok    = (const int*)  d_in[0];
    const float* cl     = (const float*)d_in[1];
    const float* plm    = (const float*)d_in[2];
    const float* ql     = (const float*)d_in[3];
    const float* si     = (const float*)d_in[4];
    const float* zij    = (const float*)d_in[5];
    const float* rl     = (const float*)d_in[6];
    const float* ln_s_w = (const float*)d_in[7];
    const float* ln_s_b = (const float*)d_in[8];
    const float* w_s    = (const float*)d_in[9];
    const float* ln_z_w = (const float*)d_in[10];
    const float* ln_z_b = (const float*)d_in[11];
    const float* w_z    = (const float*)d_in[12];
    const float* w_r    = (const float*)d_in[13];

    float* out     = (float*)d_out;
    float* cl_out  = out;                                  // 1024*128
    float* plm_out = out + 131072;                         // 1024*1024*16
    float* ql_out  = out + 131072 + 16777216;              // 1024*128

    char* ws = (char*)d_ws;
    float*    P    = (float*)ws;                           // 65536*16 f32 = 4 MB
    float*    S    = (float*)(ws + (size_t)4 * 1024 * 1024);          // 128 KB
    uint32_t* lnsi = (uint32_t*)(ws + (size_t)4 * 1024 * 1024 + 131072 * 4);

    // Token-space LN-GEMMs (WMMA, tiny, off critical path)
    pair_ln_gemm_kernel<<<1024, 128, 0, stream>>>(zij, ln_z_w, ln_z_b, w_z, P);
    ln_si_kernel<<<256, 128, 0, stream>>>(si, ln_s_w, ln_s_b, lnsi);
    s_gemm_kernel<<<32, 128, 0, stream>>>(lnsi, w_s, S);

    // Streaming gather-add (bandwidth-bound: ~128 MB HBM traffic)
    plm_add_kernel<<<16384, 256, 0, stream>>>((const float4*)plm, tok,
                                              (const float4*)P, (float4*)plm_out);
    small_out_kernel<<<512, 256, 0, stream>>>(cl, S, tok, ql, rl, w_r,
                                              cl_out, ql_out);
}
